// VASP_Encoder_60748017435306
// MI455X (gfx1250) — matrix operand order repaired and checked by reading, not compiled, not run on hardware
//
#include <hip/hip_runtime.h>
#include <hip/hip_bf16.h>
#include <math.h>

// ---------- types ----------
typedef __bf16 bf16_t;
typedef __attribute__((ext_vector_type(16))) __bf16 v16bf;
typedef __attribute__((ext_vector_type(8)))  __bf16 v8bf;
typedef __attribute__((ext_vector_type(8)))  float  v8f;
typedef __attribute__((ext_vector_type(4)))  float  v4f;

#define FEATURES 128
#define FC1      80
#define HID      65
#define G4       260      // 4*HID
#define NPAD     272      // 17 tiles of 16
#define NTILES   17
#define KPAD_H   96       // HID padded to 3*32
#define FC2      13
#define LATENT   16
#define BATCH    1024
#define SEQ      2048

// workspace layout (bytes)
#define WCOMB_ELEMS (NTILES * 4 * 512)        // 34816 bf16
#define RFRAG_ELEMS (NTILES * 3 * 512)        // 26112 bf16
#define OFF_WCOMB   0
#define OFF_RFRAG   (WCOMB_ELEMS * 2)         // 69632
#define OFF_BCOMB   (OFF_RFRAG + RFRAG_ELEMS * 2)  // 121856

// ---------- fragment helpers ----------
__device__ __forceinline__ v16bf cat8(v8bf lo, v8bf hi) {
    union { v16bf v; v8bf h[2]; } u;
    u.h[0] = lo; u.h[1] = hi;
    return u.v;
}

// A-matrix fragment (16-bit, 16x32) from row-major LDS buffer [16][stride].
// Layout: lanes 0-15 hold M=lane, K = kc*32 + {0..15};
//         lanes 16-31 hold M=lane-16, K = kc*32 + {16..31}.
__device__ __forceinline__ v16bf load_afrag(const bf16_t* buf, int stride, int kc, int lane) {
    int m  = lane & 15;
    int k0 = kc * 32 + ((lane & 16) ? 16 : 0);
    const bf16_t* p = buf + m * stride + k0;
    return cat8(*(const v8bf*)p, *(const v8bf*)(p + 8));
}

// B-matrix fragment: pre-swizzled 512-element block; lane l reads 16
// consecutive bf16 (32B) -> two global_load_b128 (L2-resident weights).
__device__ __forceinline__ v16bf load_bfrag(const bf16_t* frag, int lane) {
    const bf16_t* p = frag + lane * 16;
    return cat8(*(const v8bf*)p, *(const v8bf*)(p + 8));
}

// ---------- fast transcendentals (serial critical path of the scan) ----------
__device__ __forceinline__ float fast_rcp(float x) {
#if __has_builtin(__builtin_amdgcn_rcpf)
    return __builtin_amdgcn_rcpf(x);       // v_rcp_f32
#else
    return 1.0f / x;
#endif
}
__device__ __forceinline__ float fast_sig(float x) {
    return fast_rcp(1.0f + __expf(-x));    // v_exp_f32 + v_rcp_f32
}
__device__ __forceinline__ float fast_tanh(float x) {
#if __has_builtin(__builtin_amdgcn_tanhf)
    return __builtin_amdgcn_tanhf(x);      // v_tanh_f32 (CDNA5 TRANS op)
#else
    float e = __expf(2.0f * x);
    return 1.0f - 2.0f * fast_rcp(e + 1.0f);
#endif
}

// ---------- prep: fold fc1 into LSTM input projection, swizzle weights ----------
__global__ void vasp_prep(const float* __restrict__ W_fc1, const float* __restrict__ b_fc1,
                          const float* __restrict__ K_lstm, const float* __restrict__ R_lstm,
                          const float* __restrict__ b_lstm,
                          bf16_t* __restrict__ wcombFrag, bf16_t* __restrict__ rFrag,
                          float* __restrict__ bcomb) {
    int tid = threadIdx.x;
    // W_comb = W_fc1 @ K_lstm written directly in B-fragment layout:
    // fragment (nt,kc); element e = lane*16 + j; n = nt*16 + (lane&15);
    // k = kc*32 + ((lane>=16)?16:0) + j
    for (int e = tid; e < WCOMB_ELEMS; e += 256) {
        int frag = e >> 9;
        int idx  = e & 511;
        int nt = frag >> 2, kc = frag & 3;
        int lane = idx >> 4, j = idx & 15;
        int n = nt * 16 + (lane & 15);
        int k = kc * 32 + ((lane & 16) ? 16 : 0) + j;   // k in [0,128)
        float s = 0.0f;
        if (n < G4) {
            for (int q = 0; q < FC1; ++q)
                s += W_fc1[k * FC1 + q] * K_lstm[q * G4 + n];
        }
        wcombFrag[e] = (bf16_t)s;
    }
    // R fragments, K padded 65 -> 96 with zeros
    for (int e = tid; e < RFRAG_ELEMS; e += 256) {
        int frag = e >> 9;
        int idx  = e & 511;
        int nt = frag / 3, kc = frag % 3;
        int lane = idx >> 4, j = idx & 15;
        int n = nt * 16 + (lane & 15);
        int k = kc * 32 + ((lane & 16) ? 16 : 0) + j;
        float v = (n < G4 && k < HID) ? R_lstm[k * G4 + n] : 0.0f;
        rFrag[e] = (bf16_t)v;
    }
    // b_comb = b_fc1 @ K_lstm + b_lstm (padded to 272)
    for (int n = tid; n < NPAD; n += 256) {
        float s = 0.0f;
        if (n < G4) {
            s = b_lstm[n];
            for (int q = 0; q < FC1; ++q) s += b_fc1[q] * K_lstm[q * G4 + n];
        }
        bcomb[n] = s;
    }
}

// ---------- main: fused fc1+input-proj GEMM, LSTM scan, VAE heads ----------
__global__ __launch_bounds__(256)
void vasp_main(const float* __restrict__ x,
               const bf16_t* __restrict__ wcombFrag,
               const bf16_t* __restrict__ rFrag,
               const float* __restrict__ bcomb,
               const float* __restrict__ W_fc2, const float* __restrict__ b_fc2,
               const float* __restrict__ W_mean, const float* __restrict__ b_mean,
               const float* __restrict__ W_logvar, const float* __restrict__ b_logvar,
               const float* __restrict__ eps,
               float* __restrict__ out_mean, float* __restrict__ out_logvar,
               float* __restrict__ out_z) {
    __shared__ bf16_t xbuf[16 * FEATURES];   // bf16 A-tile of x_t              (4 KB)
    __shared__ bf16_t hbuf[16 * KPAD_H];     // bf16 A-tile of h (zero-pad K)   (3 KB)
    __shared__ float  zbuf[16 * NPAD];       // pre-activation gates            (17.4 KB)
    __shared__ float  hf[16 * 68];           // f32 h (for head)                (4.3 KB)
    __shared__ float  cf[16 * 68];           // f32 cell state                  (4.3 KB)
    __shared__ float  sbias[NPAD];
    __shared__ float  fc2buf[16 * FC2];

    const int tid  = threadIdx.x;
    const int lane = tid & 31;
    const int wave = tid >> 5;
    const int m0   = blockIdx.x * 16;        // batch-row tile

    for (int i = tid; i < 16 * KPAD_H; i += 256) hbuf[i] = (bf16_t)0.0f;
    for (int i = tid; i < 16 * 68; i += 256) { hf[i] = 0.0f; cf[i] = 0.0f; }
    for (int i = tid; i < NPAD; i += 256) sbias[i] = bcomb[i];

    // software-pipelined x staging: each thread owns 8 contiguous floats of
    // the 16x128 tile; registers hold x_t while LDS holds x_(t-1)'s bf16 tile.
    const int e0   = tid * 8;
    const int xrow = e0 >> 7;
    const int xcol = e0 & 127;
    const float* xsrc = x + (size_t)(m0 + xrow) * (SEQ * FEATURES) + xcol;

    v4f xa = __builtin_nontemporal_load((const v4f*)(xsrc));
    v4f xb = __builtin_nontemporal_load((const v4f*)(xsrc + 4));
    __syncthreads();

    for (int t = 0; t < SEQ; ++t) {
        // --- commit x_t registers -> bf16 LDS tile ---
        {
            bf16_t* dst = xbuf + xrow * FEATURES + xcol;
            dst[0] = (bf16_t)xa.x; dst[1] = (bf16_t)xa.y;
            dst[2] = (bf16_t)xa.z; dst[3] = (bf16_t)xa.w;
            dst[4] = (bf16_t)xb.x; dst[5] = (bf16_t)xb.y;
            dst[6] = (bf16_t)xb.z; dst[7] = (bf16_t)xb.w;
        }
        __syncthreads();

        // --- issue x_(t+1) loads; they fly under the WMMA + gate phases ---
        if (t + 1 < SEQ) {
            const float* nsrc = xsrc + (size_t)(t + 1) * FEATURES;
            xa = __builtin_nontemporal_load((const v4f*)(nsrc));
            xb = __builtin_nontemporal_load((const v4f*)(nsrc + 4));
            if (t + 2 < SEQ) __builtin_prefetch(nsrc + FEATURES, 0, 0);
        }

        // --- z = b_comb + x_t @ W_comb + h @ R   (bf16 WMMA, f32 accum) ---
        for (int nt = wave; nt < NTILES; nt += 8) {
            int col = nt * 16 + (lane & 15);
            float bv = sbias[col];
            v8f acc;
            #pragma unroll
            for (int r = 0; r < 8; ++r) acc[r] = bv;

            #pragma unroll
            for (int kc = 0; kc < 4; ++kc) {          // K = 128
                v16bf a = load_afrag(xbuf, FEATURES, kc, lane);
                v16bf b = load_bfrag(wcombFrag + (nt * 4 + kc) * 512, lane);
                acc = __builtin_amdgcn_wmma_f32_16x16x32_bf16(
                          false, a, false, b, (short)0, acc, false, false);
            }
            #pragma unroll
            for (int kc = 0; kc < 3; ++kc) {          // K = 96 (h zero-padded)
                v16bf a = load_afrag(hbuf, KPAD_H, kc, lane);
                v16bf b = load_bfrag(rFrag + (nt * 3 + kc) * 512, lane);
                acc = __builtin_amdgcn_wmma_f32_16x16x32_bf16(
                          false, a, false, b, (short)0, acc, false, false);
            }
            // C/D layout: vgpr r -> M = r + 8*(lane/16); N = lane%16
            int rbase = (lane & 16) ? 8 : 0;
            #pragma unroll
            for (int r = 0; r < 8; ++r)
                zbuf[(rbase + r) * NPAD + col] = acc[r];
        }
        __syncthreads();

        // --- gates + state update (native TRANS ops on the critical path) ---
        for (int idx = tid; idx < 16 * HID; idx += 256) {
            int row = idx / HID;
            int j   = idx - row * HID;
            const float* zr = zbuf + row * NPAD;
            float ig = fast_sig(zr[j]);
            float fg = fast_sig(zr[HID + j]);
            float gg = fast_tanh(zr[2 * HID + j]);
            float og = fast_sig(zr[3 * HID + j]);
            float c  = fg * cf[row * 68 + j] + ig * gg;
            float h  = og * fast_tanh(c);
            cf[row * 68 + j] = c;
            hf[row * 68 + j] = h;
            hbuf[row * KPAD_H + j] = (bf16_t)h;
        }
        __syncthreads();
    }

    // --- heads: fc2(13), z_mean/z_logvar(16), reparameterize ---
    for (int idx = tid; idx < 16 * FC2; idx += 256) {
        int row = idx / FC2, k = idx - row * FC2;
        float s = b_fc2[k];
        for (int j = 0; j < HID; ++j) s += hf[row * 68 + j] * W_fc2[j * FC2 + k];
        fc2buf[row * FC2 + k] = s;
    }
    __syncthreads();
    for (int idx = tid; idx < 16 * LATENT; idx += 256) {
        int row = idx >> 4, l = idx & 15;
        float m  = b_mean[l];
        float lv = b_logvar[l];
        for (int k = 0; k < FC2; ++k) {
            float f = fc2buf[row * FC2 + k];
            m  += f * W_mean[k * LATENT + l];
            lv += f * W_logvar[k * LATENT + l];
        }
        int gr = m0 + row;
        float e = eps[gr * LATENT + l];
        out_mean[gr * LATENT + l]   = m;
        out_logvar[gr * LATENT + l] = lv;
        out_z[gr * LATENT + l]      = m + sqrtf(__expf(lv)) * e;
    }
}

// ---------- launch ----------
extern "C" void kernel_launch(void* const* d_in, const int* in_sizes, int n_in,
                              void* d_out, int out_size, void* d_ws, size_t ws_size,
                              hipStream_t stream) {
    const float* x        = (const float*)d_in[0];
    const float* W_fc1    = (const float*)d_in[1];
    const float* b_fc1    = (const float*)d_in[2];
    const float* K_lstm   = (const float*)d_in[3];
    const float* R_lstm   = (const float*)d_in[4];
    const float* b_lstm   = (const float*)d_in[5];
    const float* W_fc2    = (const float*)d_in[6];
    const float* b_fc2    = (const float*)d_in[7];
    const float* W_mean   = (const float*)d_in[8];
    const float* b_mean   = (const float*)d_in[9];
    const float* W_logvar = (const float*)d_in[10];
    const float* b_logvar = (const float*)d_in[11];
    const float* eps      = (const float*)d_in[12];

    bf16_t* wcombFrag = (bf16_t*)((char*)d_ws + OFF_WCOMB);
    bf16_t* rFrag     = (bf16_t*)((char*)d_ws + OFF_RFRAG);
    float*  bcomb     = (float*) ((char*)d_ws + OFF_BCOMB);

    float* out_mean   = (float*)d_out;
    float* out_logvar = out_mean + BATCH * LATENT;
    float* out_z      = out_logvar + BATCH * LATENT;

    vasp_prep<<<1, 256, 0, stream>>>(W_fc1, b_fc1, K_lstm, R_lstm, b_lstm,
                                     wcombFrag, rFrag, bcomb);
    vasp_main<<<BATCH / 16, 256, 0, stream>>>(x, wcombFrag, rFrag, bcomb,
                                              W_fc2, b_fc2, W_mean, b_mean,
                                              W_logvar, b_logvar, eps,
                                              out_mean, out_logvar, out_z);
}
